// BiLSTM_CRF_61795989454974
// MI455X (gfx1250) — matrix operand order repaired and checked by reading, not compiled, not run on hardware
//
#include <hip/hip_runtime.h>
#include <hip/hip_bf16.h>
#include <math.h>

// ---------------- sizes ----------------
#define T_LEN 2048
#define EMB   256
#define HID   512
#define HD    256        // per-direction hidden
#define G4H   1024       // 4*HD gates
#define N_TAGS 7
#define TAG_START 5
#define TAG_END   6
#define NEGV (-10000.0f)

typedef _Float16 half8  __attribute__((ext_vector_type(8)));
typedef _Float16 half16 __attribute__((ext_vector_type(16)));
typedef float    f32x8  __attribute__((ext_vector_type(8)));

// ---------------------------------------------------------------------------
// Kernel 1: gather embeddings -> f16, convert W_ih and W_hh (both dirs) -> f16,
//           fold biases (b_ih + b_hh) per direction.
// ---------------------------------------------------------------------------
__global__ __launch_bounds__(256)
void prep(const int* __restrict__ words, const float* __restrict__ embed,
          const float* __restrict__ WihF, const float* __restrict__ WihB,
          const float* __restrict__ WhhF, const float* __restrict__ WhhB,
          const float* __restrict__ bihF, const float* __restrict__ bhhF,
          const float* __restrict__ bihB, const float* __restrict__ bhhB,
          _Float16* __restrict__ xs16, _Float16* __restrict__ W16,
          _Float16* __restrict__ Whh16, float* __restrict__ bias)
{
    const int XS = T_LEN * EMB;          // 524288
    const int WN = G4H * EMB;            // 262144 per direction (ih) / (hh)
    const int total = XS + 4 * WN + 2 * G4H;
    for (int idx = blockIdx.x * blockDim.x + threadIdx.x; idx < total;
         idx += gridDim.x * blockDim.x) {
        if (idx < XS) {
            int t = idx >> 8, e = idx & 255;
            xs16[idx] = (_Float16)embed[(size_t)words[t] * EMB + e];
        } else if (idx < XS + WN) {
            int i = idx - XS;
            W16[i] = (_Float16)WihF[i];
        } else if (idx < XS + 2 * WN) {
            int i = idx - XS - WN;
            W16[WN + i] = (_Float16)WihB[i];
        } else if (idx < XS + 3 * WN) {
            int i = idx - XS - 2 * WN;
            Whh16[i] = (_Float16)WhhF[i];
        } else if (idx < XS + 4 * WN) {
            int i = idx - XS - 3 * WN;
            Whh16[WN + i] = (_Float16)WhhB[i];
        } else {
            int i = idx - XS - 4 * WN;
            if (i < G4H) bias[i] = bihF[i]       + bhhF[i];
            else         bias[i] = bihB[i - G4H] + bhhB[i - G4H];
        }
    }
}

// ---------------------------------------------------------------------------
// Kernel 2: G_pre[dir][t][4H] = xs_f16 @ Wih_dir^T + (b_ih + b_hh)
// WMMA f32_16x16x32_f16.  One 16x16 output tile per wave, K = 256 = 8 steps.
//   A (16x32 f16): lane half h holds K = 8h+[0..7] and K = 16+8h+[0..7]
//   B (32x16 f16): lane half h holds K = 8h+[0..7] and K = 16+8h+[0..7]
//   C/D (16x16 f32): col = lane&15, row = v + 8*(lane>>4)
// ---------------------------------------------------------------------------
__global__ __launch_bounds__(128)
void gemm_pre(const _Float16* __restrict__ xs16, const _Float16* __restrict__ W16,
              const float* __restrict__ bias, float* __restrict__ Gpre)
{
    const int lane    = threadIdx.x & 31;
    const int wave    = threadIdx.x >> 5;
    const int half_id = lane >> 4;
    const int lrow    = lane & 15;
    const int tm  = blockIdx.x;              // time tile: 0..127
    const int tn  = blockIdx.y * 4 + wave;   // gate tile: 0..63
    const int dir = blockIdx.z;

    const _Float16* __restrict__ Arow = xs16 + (size_t)(tm * 16 + lrow) * EMB;
    const _Float16* __restrict__ Brow = W16 + (size_t)dir * G4H * EMB
                                            + (size_t)(tn * 16 + lrow) * EMB;

    f32x8 acc;
    const float bv = bias[dir * G4H + tn * 16 + lrow];   // bias indexed by column n
    #pragma unroll
    for (int i = 0; i < 8; ++i) acc[i] = bv;

    #pragma unroll
    for (int kb = 0; kb < 8; ++kb) {
        // A fragment: interleaved 8-element chunks
        const int ka = kb * 32 + half_id * 8;
        half8 alo = *(const half8*)(Arow + ka);
        half8 ahi = *(const half8*)(Arow + ka + 16);
        half16 a, b;
        // B fragment: interleaved 8-element chunks (same as A)
        const int kbb = kb * 32 + half_id * 8;
        half8 blo = *(const half8*)(Brow + kbb);
        half8 bhi = *(const half8*)(Brow + kbb + 16);
        #pragma unroll
        for (int i = 0; i < 8; ++i) {
            a[i] = alo[i]; a[8 + i] = ahi[i];
            b[i] = blo[i]; b[8 + i] = bhi[i];
        }
        acc = __builtin_amdgcn_wmma_f32_16x16x32_f16(
                  /*neg_a=*/false, a, /*neg_b=*/false, b,
                  /*c_mod=*/(short)0, acc, /*reuse_a=*/false, /*reuse_b=*/false);
    }

    float* __restrict__ out = Gpre + ((size_t)dir * T_LEN + tm * 16) * G4H
                                   + tn * 16 + lrow;
    #pragma unroll
    for (int v = 0; v < 8; ++v)
        out[(size_t)(v + half_id * 8) * G4H] = acc[v];
}

// ---------------------------------------------------------------------------
// Kernel 3: sequential LSTM recurrence, one block per direction.
// 1024 threads: thread j computes gate pre-activation j via a 256-length dot
// of f16 W_hh row j (halved L2 traffic vs f32) against f16 h in LDS, with
// f32 accumulation (V_FMA_MIX_F32 pattern).  Two independent accumulator
// chains expose VOPD dual-issue.  Threads 0..255 apply nonlinearities.
// ---------------------------------------------------------------------------
__device__ __forceinline__ float sigm(float x) { return 1.0f / (1.0f + __expf(-x)); }

__global__ __launch_bounds__(1024)
void lstm_rec(const float* __restrict__ Gpre, const _Float16* __restrict__ Whh16,
              const float* __restrict__ h0, const float* __restrict__ c0,
              float* __restrict__ lstm_out)
{
    const int dir = blockIdx.x;
    const int tid = threadIdx.x;
    const _Float16* __restrict__ Whh = Whh16 + (size_t)dir * G4H * HD;  // [1024][256] f16
    const float* __restrict__ G      = Gpre + (size_t)dir * T_LEN * G4H;

    __shared__ __align__(16) _Float16 h_sh[HD];
    __shared__ float g_sh[G4H];

    if (tid < HD) h_sh[tid] = (_Float16)h0[dir * HD + tid];
    float c = (tid < HD) ? c0[dir * HD + tid] : 0.0f;
    __syncthreads();

    const half8* __restrict__ wrow = (const half8*)(Whh + (size_t)tid * HD);

    for (int t = 0; t < T_LEN; ++t) {
        const int row = dir ? (T_LEN - 1 - t) : t;
        // prefetch next step's G_pre row while this step computes
        if (t + 1 < T_LEN) {
            const int nrow = dir ? (T_LEN - 2 - t) : (t + 1);
            __builtin_prefetch(&G[(size_t)nrow * G4H + tid], 0, 1);
        }
        float acc0 = G[(size_t)row * G4H + tid];
        float acc1 = 0.0f;
        const half8* hp = (const half8*)h_sh;
        #pragma unroll 4
        for (int k = 0; k < HD / 8; ++k) {
            half8 w  = wrow[k];
            half8 h8 = hp[k];          // LDS broadcast: all lanes read same addr
            acc0 = fmaf((float)w[0], (float)h8[0], acc0);
            acc1 = fmaf((float)w[1], (float)h8[1], acc1);
            acc0 = fmaf((float)w[2], (float)h8[2], acc0);
            acc1 = fmaf((float)w[3], (float)h8[3], acc1);
            acc0 = fmaf((float)w[4], (float)h8[4], acc0);
            acc1 = fmaf((float)w[5], (float)h8[5], acc1);
            acc0 = fmaf((float)w[6], (float)h8[6], acc0);
            acc1 = fmaf((float)w[7], (float)h8[7], acc1);
        }
        g_sh[tid] = acc0 + acc1;
        __syncthreads();
        if (tid < HD) {
            float gi = sigm(g_sh[tid]);
            float gf = sigm(g_sh[HD + tid]);
            float gg = tanhf(g_sh[2 * HD + tid]);
            float go = sigm(g_sh[3 * HD + tid]);
            c = gf * c + gi * gg;
            float h = go * tanhf(c);
            h_sh[tid] = (_Float16)h;
            lstm_out[(size_t)row * HID + dir * HD + tid] = h;
        }
        __syncthreads();
    }
}

// ---------------------------------------------------------------------------
// Kernel 4: feats[t][j] = lstm_out[t] . W_out[j] + b_out[j]
// ---------------------------------------------------------------------------
__global__ __launch_bounds__(256)
void proj(const float* __restrict__ lstm_out, const float* __restrict__ Wout,
          const float* __restrict__ bout, float* __restrict__ feats)
{
    int e = blockIdx.x * blockDim.x + threadIdx.x;
    if (e >= T_LEN * N_TAGS) return;
    int t = e / N_TAGS, j = e % N_TAGS;
    const float* __restrict__ x = lstm_out + (size_t)t * HID;
    const float* __restrict__ w = Wout + (size_t)j * HID;
    float acc = bout[j];
    #pragma unroll 4
    for (int k = 0; k < HID; k += 4)
        acc += x[k] * w[k] + x[k + 1] * w[k + 1] + x[k + 2] * w[k + 2] + x[k + 3] * w[k + 3];
    feats[e] = acc;
}

// ---------------------------------------------------------------------------
// Kernel 5: Viterbi max-plus scan + backtrace. One block, one wave.
// d_out[0] = path_score ; d_out[1 + t] = (float)path[t]
// ---------------------------------------------------------------------------
__global__ __launch_bounds__(32)
void viterbi(const float* __restrict__ feats, const float* __restrict__ trans,
             float* __restrict__ d_out)
{
    __shared__ float fv[N_TAGS];
    __shared__ float tr[N_TAGS * N_TAGS];
    __shared__ unsigned char bp_sh[T_LEN * N_TAGS];   // ~14 KB LDS
    const int tid = threadIdx.x;

    if (tid < N_TAGS * N_TAGS) tr[tid] = trans[tid];
    if (tid < N_TAGS) fv[tid] = (tid == TAG_START) ? 0.0f : NEGV;
    __syncthreads();

    for (int t = 0; t < T_LEN; ++t) {
        float best = 0.0f; int bi = 0;
        if (tid < N_TAGS) {
            best = fv[0] + tr[tid * N_TAGS];
            bi = 0;
            for (int i = 1; i < N_TAGS; ++i) {
                float v = fv[i] + tr[tid * N_TAGS + i];
                if (v > best) { best = v; bi = i; }   // strict '>' keeps first max (argmax semantics)
            }
            bp_sh[t * N_TAGS + tid] = (unsigned char)bi;
        }
        __syncthreads();
        if (tid < N_TAGS) fv[tid] = best + feats[t * N_TAGS + tid];
        __syncthreads();
    }

    if (tid == 0) {
        int ftag = 0;
        float bestT = fv[0] + tr[TAG_END * N_TAGS];
        for (int j = 1; j < N_TAGS; ++j) {
            float v = fv[j] + tr[TAG_END * N_TAGS + j];
            if (v > bestT) { bestT = v; ftag = j; }
        }
        d_out[0] = bestT;
        int tag = ftag;
        for (int t = T_LEN - 1; t >= 0; --t) {
            d_out[1 + t] = (float)tag;
            tag = (int)bp_sh[t * N_TAGS + tag];
        }
    }
}

// ---------------------------------------------------------------------------
// Host launch
// ---------------------------------------------------------------------------
extern "C" void kernel_launch(void* const* d_in, const int* in_sizes, int n_in,
                              void* d_out, int out_size, void* d_ws, size_t ws_size,
                              hipStream_t stream) {
    const int*   words = (const int*)  d_in[0];
    const float* embed = (const float*)d_in[1];
    const float* WihF  = (const float*)d_in[2];
    const float* WhhF  = (const float*)d_in[3];
    const float* bihF  = (const float*)d_in[4];
    const float* bhhF  = (const float*)d_in[5];
    const float* WihB  = (const float*)d_in[6];
    const float* WhhB  = (const float*)d_in[7];
    const float* bihB  = (const float*)d_in[8];
    const float* bhhB  = (const float*)d_in[9];
    const float* Wout  = (const float*)d_in[10];
    const float* bout  = (const float*)d_in[11];
    const float* trans = (const float*)d_in[12];
    const float* h0    = (const float*)d_in[13];
    const float* c0    = (const float*)d_in[14];

    char* ws = (char*)d_ws;
    size_t off = 0;
    auto take = [&](size_t bytes) -> char* {
        char* p = ws + off;
        off = (off + bytes + 255) & ~(size_t)255;
        return p;
    };
    _Float16* xs16  = (_Float16*)take((size_t)T_LEN * EMB * 2);        // 1 MB
    _Float16* W16   = (_Float16*)take((size_t)2 * G4H * EMB * 2);      // 1 MB
    _Float16* Whh16 = (_Float16*)take((size_t)2 * G4H * HD * 2);       // 1 MB
    float*    bias  = (float*)   take((size_t)2 * G4H * 4);            // 8 KB
    float*    Gpre  = (float*)   take((size_t)2 * T_LEN * G4H * 4);    // 16 MB
    float*    lout  = (float*)   take((size_t)T_LEN * HID * 4);        // 4 MB
    float*    fts   = (float*)   take((size_t)T_LEN * N_TAGS * 4);     // 56 KB

    hipLaunchKernelGGL(prep, dim3(1024), dim3(256), 0, stream,
                       words, embed, WihF, WihB, WhhF, WhhB,
                       bihF, bhhF, bihB, bhhB,
                       xs16, W16, Whh16, bias);
    hipLaunchKernelGGL(gemm_pre, dim3(T_LEN / 16, 16, 2), dim3(128), 0, stream,
                       xs16, W16, bias, Gpre);
    hipLaunchKernelGGL(lstm_rec, dim3(2), dim3(1024), 0, stream,
                       Gpre, Whh16, h0, c0, lout);
    hipLaunchKernelGGL(proj, dim3((T_LEN * N_TAGS + 255) / 256), dim3(256), 0, stream,
                       lout, Wout, bout, fts);
    hipLaunchKernelGGL(viterbi, dim3(1), dim3(32), 0, stream,
                       fts, trans, (float*)d_out);
}
